// BlockRegional_20529943675117
// MI455X (gfx1250) — matrix operand order repaired and checked by reading, not compiled, not run on hardware
//
#include <hip/hip_runtime.h>
#include <hip/hip_bf16.h>

typedef __bf16 bf16;
typedef __attribute__((ext_vector_type(16))) __bf16 bf16x16;
typedef __attribute__((ext_vector_type(8)))  __bf16 bf16x8;
typedef __attribute__((ext_vector_type(4)))  __bf16 bf16x4;
typedef __attribute__((ext_vector_type(8)))  float  v8f;

#define B_    8
#define S_    32
#define C_    256
#define L_    512
#define H_    8
#define D_    64
#define NROWS (B_ * S_ * C_)   /* 65536 */
#define L4_   (4 * L_)         /* 2048 */

// Flip to 0 to fall back to synchronous VGPR-round-trip tile copies.
#define USE_ASYNC_COPY 1

__device__ __forceinline__ bf16x16 combine8(bf16x8 lo, bf16x8 hi) {
  bf16x16 r;
#pragma unroll
  for (int i = 0; i < 8; ++i) { r[i] = lo[i]; r[8 + i] = hi[i]; }
  return r;
}

#if USE_ASYNC_COPY
// Async 16B global->LDS copy (ASYNCcnt-tracked, no VGPR round trip).
// lds_byte_addr: wave-relative LDS byte offset (low 32 bits of generic ptr).
__device__ __forceinline__ void cp_async_b128(unsigned int lds_byte_addr,
                                              const void* gsrc) {
  asm volatile("global_load_async_to_lds_b128 %0, %1, off"
               :: "v"(lds_byte_addr), "v"(gsrc) : "memory");
}
__device__ __forceinline__ void wait_async0() {
  asm volatile("s_wait_asynccnt 0x0" ::: "memory");
}
__device__ __forceinline__ unsigned int lds_off(const void* p) {
  return (unsigned int)(unsigned long long)p;
}
#endif

// ---------------------------------------------------------------------------
// One-time weight prep: fp32 W[K][Nout] -> bf16 WT[Nout][K] (transposed).
// ---------------------------------------------------------------------------
__global__ __launch_bounds__(256) void wt_bf16t_kernel(
    const float* __restrict__ W, bf16* __restrict__ WT, int K, int Nout) {
  const size_t idx = (size_t)blockIdx.x * 256 + threadIdx.x;
  const int k = (int)(idx / (size_t)Nout);
  const int n = (int)(idx % (size_t)Nout);
  WT[(size_t)n * K + k] = (bf16)W[idx];
}

// ---------------------------------------------------------------------------
// LayerNorm (fp32 in) -> bf16 out.  One wave32 per row of 512.
// ---------------------------------------------------------------------------
__global__ __launch_bounds__(256) void ln_bf16_kernel(
    const float* __restrict__ X, const float* __restrict__ gam,
    const float* __restrict__ bet, bf16* __restrict__ out) {
  const int lane = threadIdx.x & 31;
  const int wv   = threadIdx.x >> 5;
  const size_t row = (size_t)blockIdx.x * 8 + wv;
  const float4* xr = (const float4*)(X + row * (size_t)L_);

  float4 v[4];
  float s = 0.f;
#pragma unroll
  for (int i = 0; i < 4; ++i) {
    v[i] = xr[lane + i * 32];
    s += v[i].x + v[i].y + v[i].z + v[i].w;
  }
#pragma unroll
  for (int m = 1; m < 32; m <<= 1) s += __shfl_xor(s, m, 32);
  const float mean = s * (1.0f / L_);

  float q = 0.f;
#pragma unroll
  for (int i = 0; i < 4; ++i) {
    float dx = v[i].x - mean, dy = v[i].y - mean;
    float dz = v[i].z - mean, dw = v[i].w - mean;
    q += dx * dx + dy * dy + dz * dz + dw * dw;
  }
#pragma unroll
  for (int m = 1; m < 32; m <<= 1) q += __shfl_xor(q, m, 32);
  const float rstd = rsqrtf(q * (1.0f / L_) + 1e-5f);

#pragma unroll
  for (int i = 0; i < 4; ++i) {
    const int col = (lane + i * 32) * 4;
    bf16x4 o;
    o[0] = (bf16)((v[i].x - mean) * rstd * gam[col + 0] + bet[col + 0]);
    o[1] = (bf16)((v[i].y - mean) * rstd * gam[col + 1] + bet[col + 1]);
    o[2] = (bf16)((v[i].z - mean) * rstd * gam[col + 2] + bet[col + 2]);
    o[3] = (bf16)((v[i].w - mean) * rstd * gam[col + 3] + bet[col + 3]);
    *(bf16x4*)(out + row * L_ + col) = o;
  }
}

// ---------------------------------------------------------------------------
// WMMA GEMM: out[N x Nout] = op(A_bf16[N x K] @ WT_bf16[Nout x K]^T + bias)
// Block tile 128x256, 8 waves, wave tile 64x64 (16 accumulators), K-step 32.
// Double-buffered LDS fed by async global->LDS b128 copies.
// ---------------------------------------------------------------------------
template <typename OutT, bool RELU, bool RES>
__global__ __launch_bounds__(256) void gemm_wmma_kernel(
    const bf16* __restrict__ A, const bf16* __restrict__ WT,
    const float* __restrict__ bias, const float* __restrict__ res,
    OutT* __restrict__ out, int K, int Nout) {
  __shared__ __align__(32) bf16 As[2][128 * 32];   // [row][k]
  __shared__ __align__(32) bf16 Bs[2][256 * 32];   // [col][k] (K-contig per col)

  const int tid   = threadIdx.x;
  const int lane  = tid & 31;
  const int wv    = tid >> 5;
  const int waveM = wv >> 2;        // 0..1
  const int waveN = wv & 3;         // 0..3
  const int half  = lane >> 4;      // 0..1
  const int l15   = lane & 15;
  const size_t rowBase = (size_t)blockIdx.x * 128;
  const int colBase    = blockIdx.y * 256;

  // per-thread tile-copy assignment
  const int r  = tid >> 1;               // A: 128 rows, 2 threads/row
  const int kc = (tid & 1) * 16;         // 16 bf16 (32B) each
  const bf16* aSrc = A + (rowBase + r) * (size_t)K + kc;
  const bf16* bSrc = WT + (size_t)(colBase + tid) * K;   // B: 1 col/thread, 64B

  v8f acc[4][4];
#pragma unroll
  for (int mt = 0; mt < 4; ++mt)
#pragma unroll
    for (int nt = 0; nt < 4; ++nt)
#pragma unroll
      for (int g = 0; g < 8; ++g) acc[mt][nt][g] = 0.0f;

#define ISSUE_TILE(buf, kk0)                                                   \
  do {                                                                         \
    const bf16* as_ = aSrc + (kk0);                                            \
    const bf16* bs_ = bSrc + (kk0);                                            \
    bf16* al_ = &As[(buf)][r * 32 + kc];                                       \
    bf16* bl_ = &Bs[(buf)][tid * 32];                                          \
    TILE_CP(al_, as_, bl_, bs_);                                               \
  } while (0)

#if USE_ASYNC_COPY
#define TILE_CP(al_, as_, bl_, bs_)                                            \
  do {                                                                         \
    cp_async_b128(lds_off(al_), as_);                                          \
    cp_async_b128(lds_off(al_) + 16, (const bf16*)(as_) + 8);                  \
    cp_async_b128(lds_off(bl_), bs_);                                          \
    cp_async_b128(lds_off(bl_) + 16, (const bf16*)(bs_) + 8);                  \
    cp_async_b128(lds_off(bl_) + 32, (const bf16*)(bs_) + 16);                 \
    cp_async_b128(lds_off(bl_) + 48, (const bf16*)(bs_) + 24);                 \
  } while (0)
#else
#define TILE_CP(al_, as_, bl_, bs_)                                            \
  do {                                                                         \
    *(bf16x8*)(al_)       = *(const bf16x8*)(as_);                             \
    *(bf16x8*)((al_) + 8) = *(const bf16x8*)((as_) + 8);                       \
    *(bf16x8*)(bl_)        = *(const bf16x8*)(bs_);                            \
    *(bf16x8*)((bl_) + 8)  = *(const bf16x8*)((bs_) + 8);                      \
    *(bf16x8*)((bl_) + 16) = *(const bf16x8*)((bs_) + 16);                     \
    *(bf16x8*)((bl_) + 24) = *(const bf16x8*)((bs_) + 24);                     \
  } while (0)
#endif

  ISSUE_TILE(0, 0);

  for (int k0 = 0; k0 < K; k0 += 32) {
    const int buf = (k0 >> 5) & 1;
#if USE_ASYNC_COPY
    wait_async0();                 // this wave's tile landed in LDS
#endif
    __syncthreads();               // everyone's tile visible
    if (k0 + 32 < K) ISSUE_TILE(buf ^ 1, k0 + 32);   // prefetch next tile

    bf16x16 af[4], bfr[4];
#pragma unroll
    for (int mt = 0; mt < 4; ++mt) {
      const bf16* p = &As[buf][(waveM * 64 + mt * 16 + l15) * 32 + half * 16];
      af[mt] = combine8(*(const bf16x8*)p, *(const bf16x8*)(p + 8));
    }
#pragma unroll
    for (int nt = 0; nt < 4; ++nt) {
      const bf16* p = &Bs[buf][(waveN * 64 + nt * 16 + l15) * 32 + half * 16];
      bfr[nt] = *(const bf16x16*)p;
    }
#pragma unroll
    for (int mt = 0; mt < 4; ++mt)
#pragma unroll
      for (int nt = 0; nt < 4; ++nt)
        acc[mt][nt] = __builtin_amdgcn_wmma_f32_16x16x32_bf16(
            false, af[mt], false, bfr[nt], (short)0, acc[mt][nt], false, false);

    __syncthreads();               // all reads of buf done before its refill
  }

  // epilogue: C layout -> (m = g + 8*half, n = l15) per 16x16 tile
#pragma unroll
  for (int nt = 0; nt < 4; ++nt) {
    const int col = colBase + waveN * 64 + nt * 16 + l15;
    const float bv = bias ? bias[col] : 0.0f;
#pragma unroll
    for (int mt = 0; mt < 4; ++mt) {
#pragma unroll
      for (int g = 0; g < 8; ++g) {
        const size_t row = rowBase + waveM * 64 + mt * 16 + g + half * 8;
        float vv = acc[mt][nt][g] + bv;
        if (RELU) vv = fmaxf(vv, 0.0f);
        if (RES)  vv += res[row * (size_t)Nout + col];
        out[row * (size_t)Nout + col] = (OutT)vv;
      }
    }
  }
#undef ISSUE_TILE
#undef TILE_CP
}

// ---------------------------------------------------------------------------
// Attention: one block per (b,s,head). 16 waves; wave w owns 16 query rows.
// scores = softmax( Q Kᵀ / 8 );  O = scores @ V.  All matmuls via bf16 WMMA.
// ---------------------------------------------------------------------------
__global__ __launch_bounds__(512) void attn_kernel(
    const bf16* __restrict__ Q, const bf16* __restrict__ Kp,
    const bf16* __restrict__ V, bf16* __restrict__ O) {
  extern __shared__ __align__(128) char smem[];
  bf16* Qs = (bf16*)smem;            // [256][64]
  bf16* Ks = Qs + 256 * 64;          // [256][64]
  bf16* Vt = Ks + 256 * 64;          // [64][256]  (d-major: K-contig per column)
  bf16* Ps = Vt + 64 * 256;          // 16 waves x [16][32] staging

  const int tid  = threadIdx.x;
  const int lane = tid & 31;
  const int wv   = tid >> 5;
  const int half = lane >> 4;
  const int l15  = lane & 15;

  const int head = blockIdx.x & (H_ - 1);
  const int bs   = blockIdx.x >> 3;
  const size_t rowBase = (size_t)bs * C_;
  const int colBase    = head * D_;

  {   // cooperative loads: each thread moves 32 bf16 of Q, K, V
    const int c  = tid >> 1;
    const int d0 = (tid & 1) * 32;
    const bf16* qsrc = Q + (rowBase + c) * L_ + colBase + d0;
    const bf16* ksrc = Kp + (rowBase + c) * L_ + colBase + d0;
    const bf16* vsrc = V + (rowBase + c) * L_ + colBase + d0;
#pragma unroll
    for (int i = 0; i < 4; ++i) {
      *(bf16x8*)(Qs + c * 64 + d0 + i * 8) = *(const bf16x8*)(qsrc + i * 8);
      *(bf16x8*)(Ks + c * 64 + d0 + i * 8) = *(const bf16x8*)(ksrc + i * 8);
    }
    bf16x8 vr[4];
#pragma unroll
    for (int i = 0; i < 4; ++i) vr[i] = *(const bf16x8*)(vsrc + i * 8);
#pragma unroll
    for (int i = 0; i < 4; ++i)
#pragma unroll
      for (int e = 0; e < 8; ++e)
        Vt[(d0 + i * 8 + e) * 256 + c] = vr[i][e];
  }
  __syncthreads();

  const int qbase = wv * 16;

  bf16x16 aq[2];
#pragma unroll
  for (int kk = 0; kk < 2; ++kk) {
    const bf16* p = Qs + (qbase + l15) * 64 + kk * 32 + half * 16;
    aq[kk] = combine8(*(const bf16x8*)p, *(const bf16x8*)(p + 8));
  }

  v8f sc[16];
#pragma unroll
  for (int j = 0; j < 16; ++j)
#pragma unroll
    for (int g = 0; g < 8; ++g) sc[j][g] = 0.0f;

#pragma unroll
  for (int j = 0; j < 16; ++j) {
#pragma unroll
    for (int kk = 0; kk < 2; ++kk) {
      const bf16* p = Ks + (j * 16 + l15) * 64 + kk * 32 + half * 16;
      const bf16x16 bk = *(const bf16x16*)p;
      sc[j] = __builtin_amdgcn_wmma_f32_16x16x32_bf16(
          false, aq[kk], false, bk, (short)0, sc[j], false, false);
    }
  }

  // softmax over 256 keys; row g+8*half lives in lanes of one 16-lane half
  const float scale = 0.125f;   // D^-0.5
#pragma unroll
  for (int g = 0; g < 8; ++g) {
    float mx = -1e30f;
#pragma unroll
    for (int j = 0; j < 16; ++j) mx = fmaxf(mx, sc[j][g]);
#pragma unroll
    for (int msk = 1; msk < 16; msk <<= 1) mx = fmaxf(mx, __shfl_xor(mx, msk, 32));
    float sum = 0.f;
#pragma unroll
    for (int j = 0; j < 16; ++j) {
      const float e = __expf((sc[j][g] - mx) * scale);
      sc[j][g] = e;
      sum += e;
    }
#pragma unroll
    for (int msk = 1; msk < 16; msk <<= 1) sum += __shfl_xor(sum, msk, 32);
    const float rs = 1.0f / sum;
#pragma unroll
    for (int j = 0; j < 16; ++j) sc[j][g] *= rs;
  }

  // O = P @ V ; stage P chunks (C-layout -> A-layout) via per-wave LDS
  v8f ao[4];
#pragma unroll
  for (int dt = 0; dt < 4; ++dt)
#pragma unroll
    for (int g = 0; g < 8; ++g) ao[dt][g] = 0.0f;

  bf16* Pw = Ps + wv * (16 * 32);
#pragma unroll
  for (int jc = 0; jc < 8; ++jc) {
#pragma unroll
    for (int t2 = 0; t2 < 2; ++t2) {
      const int j = jc * 2 + t2;
#pragma unroll
      for (int g = 0; g < 8; ++g)
        Pw[(g + half * 8) * 32 + t2 * 16 + l15] = (bf16)sc[j][g];
    }
    const bf16* pp = Pw + l15 * 32 + half * 16;    // same-wave LDS: in order
    const bf16x16 ap = combine8(*(const bf16x8*)pp, *(const bf16x8*)(pp + 8));
#pragma unroll
    for (int dt = 0; dt < 4; ++dt) {
      const bf16* vp = Vt + (dt * 16 + l15) * 256 + jc * 32 + half * 16;
      const bf16x16 bv = *(const bf16x16*)vp;
      ao[dt] = __builtin_amdgcn_wmma_f32_16x16x32_bf16(
          false, ap, false, bv, (short)0, ao[dt], false, false);
    }
  }

#pragma unroll
  for (int dt = 0; dt < 4; ++dt)
#pragma unroll
    for (int g = 0; g < 8; ++g) {
      const size_t row = rowBase + qbase + g + half * 8;
      O[row * L_ + colBase + dt * 16 + l15] = (bf16)ao[dt][g];
    }
}

// ---------------------------------------------------------------------------
extern "C" void kernel_launch(void* const* d_in, const int* in_sizes, int n_in,
                              void* d_out, int out_size, void* d_ws, size_t ws_size,
                              hipStream_t stream) {
  (void)in_sizes; (void)n_in; (void)out_size; (void)ws_size;
  const float* x     = (const float*)d_in[0];
  const float* ln1_g = (const float*)d_in[1];
  const float* ln1_b = (const float*)d_in[2];
  const float* Wq    = (const float*)d_in[3];
  const float* Wk    = (const float*)d_in[4];
  const float* Wv    = (const float*)d_in[5];
  const float* Wo    = (const float*)d_in[6];
  const float* bo    = (const float*)d_in[7];
  const float* ln2_g = (const float*)d_in[8];
  const float* ln2_b = (const float*)d_in[9];
  const float* W1    = (const float*)d_in[10];
  const float* b1    = (const float*)d_in[11];
  const float* W2    = (const float*)d_in[12];
  const float* b2    = (const float*)d_in[13];
  float* out = (float*)d_out;

  const size_t SZ = (size_t)NROWS * L_ * sizeof(bf16);   // 64 MiB
  char* ws = (char*)d_ws;
  bf16* hb = (bf16*)(ws);            // LN output (reused for LN2)
  bf16* qb = (bf16*)(ws + 1 * SZ);
  bf16* kb = (bf16*)(ws + 2 * SZ);
  bf16* vb = (bf16*)(ws + 3 * SZ);
  bf16* ob = (bf16*)(ws + 4 * SZ);
  bf16* tb = (bf16*)(ws + 1 * SZ);   // FFN hidden (N x 2048) reuses q/k/v/o span
  bf16* wqT = (bf16*)(ws + 5 * SZ);  // pre-transposed bf16 weights (~6 MB)
  bf16* wkT = wqT + (size_t)L_ * L_;
  bf16* wvT = wkT + (size_t)L_ * L_;
  bf16* woT = wvT + (size_t)L_ * L_;
  bf16* w1T = woT + (size_t)L_ * L_;
  bf16* w2T = w1T + (size_t)L_ * L4_;

  const int ATTN_SMEM = (256 * 64 + 256 * 64 + 64 * 256 + 16 * 16 * 32) * 2; // 112 KB
  (void)hipFuncSetAttribute((const void*)attn_kernel,
                            hipFuncAttributeMaxDynamicSharedMemorySize, ATTN_SMEM);

  // 0) one-time weight convert+transpose to bf16 (tiny)
  wt_bf16t_kernel<<<(L_ * L_) / 256, 256, 0, stream>>>(Wq, wqT, L_, L_);
  wt_bf16t_kernel<<<(L_ * L_) / 256, 256, 0, stream>>>(Wk, wkT, L_, L_);
  wt_bf16t_kernel<<<(L_ * L_) / 256, 256, 0, stream>>>(Wv, wvT, L_, L_);
  wt_bf16t_kernel<<<(L_ * L_) / 256, 256, 0, stream>>>(Wo, woT, L_, L_);
  wt_bf16t_kernel<<<(L_ * L4_) / 256, 256, 0, stream>>>(W1, w1T, L_, L4_);
  wt_bf16t_kernel<<<(L4_ * L_) / 256, 256, 0, stream>>>(W2, w2T, L4_, L_);

  // 1) LN1 -> bf16
  ln_bf16_kernel<<<NROWS / 8, 256, 0, stream>>>(x, ln1_g, ln1_b, hb);

  // 2) Q/K/V projections (bf16 out)
  dim3 g512(NROWS / 128, L_ / 256);
  gemm_wmma_kernel<bf16, false, false><<<g512, 256, 0, stream>>>(hb, wqT, nullptr, nullptr, qb, L_, L_);
  gemm_wmma_kernel<bf16, false, false><<<g512, 256, 0, stream>>>(hb, wkT, nullptr, nullptr, kb, L_, L_);
  gemm_wmma_kernel<bf16, false, false><<<g512, 256, 0, stream>>>(hb, wvT, nullptr, nullptr, vb, L_, L_);

  // 3) attention per (b,s,head)
  attn_kernel<<<B_ * S_ * H_, 512, ATTN_SMEM, stream>>>(qb, kb, vb, ob);

  // 4) x1 = x + o @ Wo + bo  (fp32 into d_out)
  gemm_wmma_kernel<float, false, true><<<g512, 256, 0, stream>>>(ob, woT, bo, x, out, L_, L_);

  // 5) LN2 -> bf16
  ln_bf16_kernel<<<NROWS / 8, 256, 0, stream>>>(out, ln2_g, ln2_b, hb);

  // 6) t = relu(h2 @ W1 + b1)  (bf16, N x 2048)
  dim3 g2048(NROWS / 128, L4_ / 256);
  gemm_wmma_kernel<bf16, true, false><<<g2048, 256, 0, stream>>>(hb, w1T, b1, nullptr, tb, L_, L4_);

  // 7) out = x1 + t @ W2 + b2
  gemm_wmma_kernel<float, false, true><<<g512, 256, 0, stream>>>(tb, w2T, b2, out, out, L4_, L_);
}
